// TransConvLayer_47957604827357
// MI455X (gfx1250) — matrix operand order repaired and checked by reading, not compiled, not run on hardware
//
#include <hip/hip_runtime.h>
#include <hip/hip_bf16.h>

// ---------------------------------------------------------------------------
// Problem constants
// ---------------------------------------------------------------------------
#define NN      65536     // nodes
#define INCH    512
#define HEADS   8
#define DCH     64
#define HD      512       // HEADS*DCH

typedef __attribute__((ext_vector_type(16))) __bf16 bf16x16;
typedef __attribute__((ext_vector_type(8)))  float  f32x8;
typedef __attribute__((ext_vector_type(4)))  unsigned int u32x4;
typedef __attribute__((ext_vector_type(8)))  int          i32x8;
typedef __attribute__((ext_vector_type(4)))  int          i32x4;

// Scheduling pins: 0x020 = VMEM read group, 0x008 = matrix (WMMA) group.
#if __has_builtin(__builtin_amdgcn_sched_group_barrier)
#define SGB(mask, size, id) __builtin_amdgcn_sched_group_barrier((mask), (size), (id))
#else
#define SGB(mask, size, id)
#endif

union Frag16 {                 // one WMMA bf16 A or B fragment (8 VGPRs)
    bf16x16        v;
    uint4          q[2];
    unsigned short u[16];
};

__device__ __forceinline__ unsigned short f2bf(float f) {
    unsigned int b = __float_as_uint(f);
    unsigned int r = b + 0x7FFFu + ((b >> 16) & 1u);   // round-to-nearest-even
    return (unsigned short)(r >> 16);
}
__device__ __forceinline__ float bf2f(unsigned short h) {
    return __uint_as_float(((unsigned int)h) << 16);
}
__device__ __forceinline__ f32x8 zero8() {
    f32x8 z;
#pragma unroll
    for (int i = 0; i < 8; ++i) z[i] = 0.0f;
    return z;
}
__device__ __forceinline__ f32x8 wmma_bf16(const bf16x16& a, const bf16x16& b, f32x8 c) {
    return __builtin_amdgcn_wmma_f32_16x16x32_bf16(false, a, false, b, (short)0, c, false, false);
}

// ---------------------------------------------------------------------------
// TDM: async 2-D tile load (bf16 elements) Global -> LDS, per ISA 08 §8.3/8.4.
// D# group0: count=1 | lds_addr | global_addr | type=2
// D# group1: data_size=1(2B) | tensor_dim0/1 | tile_dim0/1 | dim0 stride
// ---------------------------------------------------------------------------
#if __has_builtin(__builtin_amdgcn_tensor_load_to_lds)
#define HAVE_TDM 1
__device__ __forceinline__ void tdm_load_tile_2d_bf16(unsigned lds_byte_off,
                                                      const void* global_tile_addr,
                                                      unsigned tile_d0, unsigned tile_d1,
                                                      unsigned tensor_d0, unsigned tensor_d1,
                                                      unsigned stride0_elems) {
    unsigned long long ga = (unsigned long long)global_tile_addr;
    u32x4 g0;
    g0[0] = 1u;                                            // count=1 (valid user D#)
    g0[1] = lds_byte_off;                                  // lds_addr [63:32]
    g0[2] = (unsigned)(ga & 0xFFFFFFFFu);                  // global_addr lo
    g0[3] = (unsigned)((ga >> 32) & 0x01FFFFFFu) | (2u << 30); // addr hi | type=2
    i32x8 g1;
    g1[0] = (int)(1u << 16);                               // wg_mask=0, data_size=1 (2B)
    g1[1] = (int)((tensor_d0 & 0xFFFFu) << 16);            // tensor_dim0[15:0] @ bits63:48
    g1[2] = (int)(((tensor_d0 >> 16) & 0xFFFFu) | ((tensor_d1 & 0xFFFFu) << 16));
    g1[3] = (int)(((tensor_d1 >> 16) & 0xFFFFu) | ((tile_d0 & 0xFFFFu) << 16));
    g1[4] = (int)(tile_d1 & 0xFFFFu);                      // tile_dim1, tile_dim2=0
    g1[5] = (int)stride0_elems;                            // tensor_dim0_stride lo32
    g1[6] = 0;                                             // stride0 hi16 | stride1 lo16
    g1[7] = 0;
    i32x4 z4; z4[0] = z4[1] = z4[2] = z4[3] = 0;
#if defined(__clang_major__) && (__clang_major__ >= 23)
    i32x8 z8;
#pragma unroll
    for (int i = 0; i < 8; ++i) z8[i] = 0;
    __builtin_amdgcn_tensor_load_to_lds(g0, g1, z4, z4, z8, 0);
#else
    __builtin_amdgcn_tensor_load_to_lds(g0, g1, z4, z4, 0);
#endif
}
#else
#define HAVE_TDM 0
#endif

// ---------------------------------------------------------------------------
// 0) fp32 -> bf16 conversion (8 elems/thread, vectorized), accumulator zeroing
// ---------------------------------------------------------------------------
__global__ void __launch_bounds__(256) cvt_kernel(const float* __restrict__ src,
                                                  unsigned short* __restrict__ dst, int n8) {
    int i = blockIdx.x * blockDim.x + threadIdx.x;
    int stride = gridDim.x * blockDim.x;
    for (; i < n8; i += stride) {
        float4 f0 = *(const float4*)(src + (size_t)i * 8);
        float4 f1 = *(const float4*)(src + (size_t)i * 8 + 4);
        ushort pk[8];
        pk[0] = f2bf(f0.x); pk[1] = f2bf(f0.y); pk[2] = f2bf(f0.z); pk[3] = f2bf(f0.w);
        pk[4] = f2bf(f1.x); pk[5] = f2bf(f1.y); pk[6] = f2bf(f1.z); pk[7] = f2bf(f1.w);
        *(uint4*)(dst + (size_t)i * 8) = *(const uint4*)pk;
    }
}

__global__ void __launch_bounds__(256) zero_kernel(float* __restrict__ p, int n) {
    int i = blockIdx.x * blockDim.x + threadIdx.x;
    int stride = gridDim.x * blockDim.x;
    for (; i < n; i += stride) p[i] = 0.0f;
}

// ---------------------------------------------------------------------------
// 1) Projection GEMM: Out[N, 512] = X[N,512] @ W^T + bias  (bf16 WMMA, fp32 acc)
//    Wave: M=16 x N=64 (4 tiles). Ping-pong register double-buffer, with the
//    schedule pinned via sched_group_barrier: interleave (2 VMEM reads, 1 WMMA)
//    so every WMMA issues behind a partial s_wait_loadcnt and ~8 loads stay
//    in flight over the matrix pipe.
// ---------------------------------------------------------------------------
__global__ void __launch_bounds__(256) proj_kernel(const unsigned short* __restrict__ X,
                                                   const unsigned short* __restrict__ W,
                                                   const float* __restrict__ bias,
                                                   unsigned short* __restrict__ Out,
                                                   float* __restrict__ sumsq_acc,
                                                   float* __restrict__ kssum_acc) {
    const int lane = threadIdx.x & 31;
    const int wave = threadIdx.x >> 5;           // 8 waves
    const int m0   = blockIdx.x * 16;
    const int o0   = wave * 64;

    const int ncol = lane & 15;
    const int sub  = (lane >> 4) << 3;           // 0 or 8; unused here (A and B frags use ksel)
    const int ksel = (lane >> 4) << 4;           // B frag: 0 or 16
    const int row  = m0 + ncol;

    const unsigned short* arow = X + (size_t)row * INCH;
    const unsigned short* wbase = W + (size_t)(o0 + ncol) * INCH + ksel;

    f32x8 acc[4];
#pragma unroll
    for (int t = 0; t < 4; ++t) acc[t] = zero8();

    auto load_set = [&](Frag16& a, Frag16 (&b)[4], int kk) {
        a.q[0] = *(const uint4*)(arow + kk + ksel);
        a.q[1] = *(const uint4*)(arow + kk + ksel + 8);
#pragma unroll
        for (int t = 0; t < 4; ++t) {
            const unsigned short* brow = wbase + (size_t)(t * 16) * INCH + kk;
            b[t].q[0] = *(const uint4*)brow;
            b[t].q[1] = *(const uint4*)(brow + 8);
        }
    };

    Frag16 aP, bP[4], aQ, bQ[4];                 // ping / pong fragment sets
    load_set(aP, bP, 0);

#pragma unroll
    for (int k0 = 0; k0 < INCH; k0 += 64) {
        load_set(aQ, bQ, k0 + 32);               // pong loads fly over ping WMMAs
#pragma unroll
        for (int t = 0; t < 4; ++t)
            acc[t] = wmma_bf16(aP.v, bP[t].v, acc[t]);
        const bool more = (k0 + 64 < INCH);
        if (more)
            load_set(aP, bP, k0 + 64);           // ping loads fly over pong WMMAs
#pragma unroll
        for (int t = 0; t < 4; ++t)
            acc[t] = wmma_bf16(aQ.v, bQ[t].v, acc[t]);

        // Pin the per-iteration schedule: 20 (or 10) loads, 8 WMMAs interleaved.
        if (more) {
#pragma unroll
            for (int g = 0; g < 8; ++g) {
                SGB(0x020, 2, 0);                // 2 VMEM reads
                SGB(0x008, 1, 0);                // 1 WMMA
            }
            SGB(0x020, 4, 0);                    // trailing 4 VMEM reads
        } else {
#pragma unroll
            for (int g = 0; g < 5; ++g) {
                SGB(0x020, 2, 0);
                SGB(0x008, 1, 0);
            }
            SGB(0x008, 3, 0);                    // final 3 WMMAs
        }
    }

    // Epilogue: bias, store bf16, reductions
    const int mhalf = (lane >> 4) * 8;
    float ssq = 0.0f;
    float colsum[4] = {0.f, 0.f, 0.f, 0.f};
#pragma unroll
    for (int t = 0; t < 4; ++t) {
        const int oc = o0 + t * 16 + ncol;
        const float bv = bias[oc];
#pragma unroll
        for (int r = 0; r < 8; ++r) {
            float v = acc[t][r] + bv;
            Out[(size_t)(m0 + mhalf + r) * HD + oc] = f2bf(v);
            ssq += v * v;
            colsum[t] += v;
        }
    }
    if (sumsq_acc) {
        ssq += __shfl_xor(ssq, 1, 32);
        ssq += __shfl_xor(ssq, 2, 32);
        ssq += __shfl_xor(ssq, 4, 32);
        ssq += __shfl_xor(ssq, 8, 32);
        ssq += __shfl_xor(ssq, 16, 32);
        if (lane == 0) atomicAdd(sumsq_acc, ssq);
    }
    if (kssum_acc) {
#pragma unroll
        for (int t = 0; t < 4; ++t) {
            float cs = colsum[t] + __shfl_xor(colsum[t], 16, 32);
            if (lane < 16) atomicAdd(&kssum_acc[o0 + t * 16 + lane], cs);
        }
    }
}

// ---------------------------------------------------------------------------
// 2) kvs[h][m][d] = sum_l ks[l, h*64+m] * vs[l, h*64+d]   (split-L)
//    Double-buffered TDM pipeline: wave 0 issues tensor_load_to_lds for tile
//    i+1 into the alternate LDS buffer, then waits tensorcnt<=2 (tile i done,
//    next tile still in flight; TDM ops of one wave complete in order).
//    grid = (64 L-chunks, 8 heads); block = 256 (8 waves). Wave owns 2 tiles.
// ---------------------------------------------------------------------------
__global__ void __launch_bounds__(256) kvs_kernel(const unsigned short* __restrict__ KSb,
                                                  const unsigned short* __restrict__ VSb,
                                                  float* __restrict__ kvs) {
    __shared__ __align__(16) unsigned short lk[2][32 * 64];
    __shared__ __align__(16) unsigned short lv[2][32 * 64];

    const int h    = blockIdx.y;
    const int lane = threadIdx.x & 31;
    const int wave = threadIdx.x >> 5;
    const int mt   = wave & 3;                   // m tile 0..3
    const int ntb  = (wave >> 2) * 2;            // n tiles ntb, ntb+1

    const int ncol = lane & 15;
    const int sub  = (lane >> 4) << 3;
    const int ksel = (lane >> 4) << 4;

    f32x8 acc0 = zero8(), acc1 = zero8();
    const int l0base = blockIdx.x * 1024;
    const int NT = 32;                           // 32 tiles of 32 rows

#if HAVE_TDM
    unsigned lkoff[2] = { (unsigned)(size_t)&lk[0][0], (unsigned)(size_t)&lk[1][0] };
    unsigned lvoff[2] = { (unsigned)(size_t)&lv[0][0], (unsigned)(size_t)&lv[1][0] };
    if (wave == 0) {
        const size_t g0 = (size_t)l0base * HD + h * DCH;
        tdm_load_tile_2d_bf16(lkoff[0], KSb + g0, 64, 32, HD, NN, HD);
        tdm_load_tile_2d_bf16(lvoff[0], VSb + g0, 64, 32, HD, NN, HD);
    }
#else
    const int lr = threadIdx.x >> 3;             // 0..31 (staging row)
    const int lc = (threadIdx.x & 7) * 8;        // staging col start
#endif

    for (int i = 0; i < NT; ++i) {
        const int cur = i & 1;
#if HAVE_TDM
        if (wave == 0) {
            if (i + 1 < NT) {
                const size_t gn = (size_t)(l0base + (i + 1) * 32) * HD + h * DCH;
                tdm_load_tile_2d_bf16(lkoff[cur ^ 1], KSb + gn, 64, 32, HD, NN, HD);
                tdm_load_tile_2d_bf16(lvoff[cur ^ 1], VSb + gn, 64, 32, HD, NN, HD);
                __builtin_amdgcn_s_wait_tensorcnt(2);   // tile i complete
            } else {
                __builtin_amdgcn_s_wait_tensorcnt(0);
            }
        }
        __syncthreads();                         // release consumers of buf[cur]
#else
        __syncthreads();
        {
            const size_t gl = (size_t)(l0base + i * 32 + lr) * HD + h * DCH + lc;
            *(uint4*)&lk[cur][lr * 64 + lc] = *(const uint4*)(KSb + gl);
            *(uint4*)&lv[cur][lr * 64 + lc] = *(const uint4*)(VSb + gl);
        }
        __syncthreads();
#endif
        Frag16 a, b0, b1;
        const int m = mt * 16 + ncol;
#pragma unroll
        for (int j = 0; j < 8; ++j) {
            a.u[j]     = lk[cur][(ksel + j) * 64 + m];
            a.u[8 + j] = lk[cur][(ksel + 8 + j) * 64 + m];
        }
        const int d0 = ntb * 16 + ncol;
#pragma unroll
        for (int j = 0; j < 16; ++j) {
            b0.u[j] = lv[cur][(ksel + j) * 64 + d0];
            b1.u[j] = lv[cur][(ksel + j) * 64 + d0 + 16];
        }
        acc0 = wmma_bf16(a.v, b0.v, acc0);
        acc1 = wmma_bf16(a.v, b1.v, acc1);
#if HAVE_TDM
        __syncthreads();                         // all reads of buf[cur] done before overwrite
#endif
    }

    const int hi8 = (lane >> 4) * 8;
#pragma unroll
    for (int r = 0; r < 8; ++r) {
        const int m = mt * 16 + r + hi8;
        atomicAdd(&kvs[h * 4096 + m * 64 + ntb * 16 + ncol], acc0[r]);
        atomicAdd(&kvs[h * 4096 + m * 64 + (ntb + 1) * 16 + ncol], acc1[r]);
    }
}

// ---------------------------------------------------------------------------
// 3) attention epilogue:
//    num  = scale*(qs @ kvs) + N*vs ;  norm = scale*(qs . ks_sum) + N
//    out[n,d] = mean_h num/norm.    grid = 512 blocks x 128 nodes, wave = head.
// ---------------------------------------------------------------------------
__global__ void __launch_bounds__(256) attn_kernel(const unsigned short* __restrict__ QSb,
                                                   const unsigned short* __restrict__ VSb,
                                                   const float* __restrict__ kvs,
                                                   const float* __restrict__ kssum,
                                                   const float* __restrict__ sumsq,
                                                   float* __restrict__ out) {
    __shared__ float lds_acc[16 * 64];

    const int lane = threadIdx.x & 31;
    const int h    = threadIdx.x >> 5;           // wave == head
    const int ncol = lane & 15;
    const int sub  = (lane >> 4) << 3;
    const int ksel = (lane >> 4) << 4;

    const float scale = rsqrtf(sumsq[0]) * rsqrtf(sumsq[1]);
    const float Nf = (float)NN;

    // Preload kvs[h] as B fragments (4 n-tiles x 2 k-steps) and fragment-aligned ks_sum
    Frag16 bfr[4][2];
#pragma unroll
    for (int nt = 0; nt < 4; ++nt) {
        const int d = nt * 16 + ncol;
#pragma unroll
        for (int ks = 0; ks < 2; ++ks)
#pragma unroll
            for (int j = 0; j < 16; ++j)
                bfr[nt][ks].u[j] = f2bf(kvs[h * 4096 + (ks * 32 + ((j < 8) ? (sub + j) : (16 + sub + (j - 8)))) * 64 + d]);
    }
    float kssA[2][16];
#pragma unroll
    for (int ks = 0; ks < 2; ++ks)
#pragma unroll
        for (int j = 0; j < 16; ++j) {
            const int kk = ks * 32 + ((j < 8) ? (sub + j) : (16 + sub + (j - 8)));
            kssA[ks][j] = kssum[h * 64 + kk];
        }

    const int mbase = blockIdx.x * 128;
    for (int mi = 0; mi < 8; ++mi) {
        for (int i = threadIdx.x; i < 1024; i += 256) lds_acc[i] = 0.0f;
        __syncthreads();

        const int m0  = mbase + mi * 16;
        const unsigned short* qrow = QSb + (size_t)(m0 + ncol) * HD + h * DCH;
        Frag16 a0, a1;
        a0.q[0] = *(const uint4*)(qrow + sub);
        a0.q[1] = *(const uint4*)(qrow + 16 + sub);
        a1.q[0] = *(const uint4*)(qrow + 32 + sub);
        a1.q[1] = *(const uint4*)(qrow + 48 + sub);

        // normalizer dot: qs[node] . ks_sum[h]
        float dotp = 0.0f;
#pragma unroll
        for (int j = 0; j < 16; ++j)
            dotp += bf2f(a0.u[j]) * kssA[0][j] + bf2f(a1.u[j]) * kssA[1][j];
        dotp += __shfl_xor(dotp, 16, 32);        // lanes 0..15 now hold node ncol's dot

        f32x8 acc[4];
#pragma unroll
        for (int nt = 0; nt < 4; ++nt) {
            acc[nt] = zero8();
            acc[nt] = wmma_bf16(a0.v, bfr[nt][0].v, acc[nt]);
            acc[nt] = wmma_bf16(a1.v, bfr[nt][1].v, acc[nt]);
        }

        const int hi8 = (lane >> 4) * 8;
#pragma unroll
        for (int nt = 0; nt < 4; ++nt) {
            const int d = nt * 16 + ncol;
#pragma unroll
            for (int r = 0; r < 8; ++r) {
                const int node = r + hi8;
                const float nrm = scale * __shfl(dotp, node, 32) + Nf;
                const float vsv = bf2f(VSb[(size_t)(m0 + node) * HD + h * DCH + d]);
                const float num = scale * acc[nt][r] + Nf * vsv;
                atomicAdd(&lds_acc[node * 64 + d], num / nrm);
            }
        }
        __syncthreads();
        for (int i = threadIdx.x; i < 1024; i += 256) {
            const int node = i >> 6, d = i & 63;
            out[(size_t)(m0 + node) * DCH + d] = lds_acc[i] * 0.125f;
        }
        __syncthreads();
    }
}

// ---------------------------------------------------------------------------
// Host launcher
// ---------------------------------------------------------------------------
extern "C" void kernel_launch(void* const* d_in, const int* in_sizes, int n_in,
                              void* d_out, int out_size, void* d_ws, size_t ws_size,
                              hipStream_t stream) {
    const float* q_in  = (const float*)d_in[0];
    const float* s_in  = (const float*)d_in[1];
    const float* Wq_w  = (const float*)d_in[2];
    const float* Wq_b  = (const float*)d_in[3];
    const float* Wk_w  = (const float*)d_in[4];
    const float* Wk_b  = (const float*)d_in[5];
    const float* Wv_w  = (const float*)d_in[6];
    const float* Wv_b  = (const float*)d_in[7];

    char* base = (char*)d_ws;
    const size_t SZ_NHD = (size_t)NN * HD * sizeof(unsigned short);   // 64 MB
    const size_t SZ_W   = (size_t)HD * INCH * sizeof(unsigned short); // 512 KB

    unsigned short* QS  = (unsigned short*)(base);
    unsigned short* KS  = (unsigned short*)(base + SZ_NHD);
    unsigned short* VS  = (unsigned short*)(base + 2 * SZ_NHD);
    unsigned short* XQ  = (unsigned short*)(base + 3 * SZ_NHD);
    unsigned short* XS  = (unsigned short*)(base + 4 * SZ_NHD);
    unsigned short* WQB = (unsigned short*)(base + 5 * SZ_NHD);
    unsigned short* WKB = (unsigned short*)(base + 5 * SZ_NHD + SZ_W);
    unsigned short* WVB = (unsigned short*)(base + 5 * SZ_NHD + 2 * SZ_W);
    float* KVS   = (float*)(base + 5 * SZ_NHD + 3 * SZ_W);            // 8*64*64
    float* KSSUM = KVS + HEADS * DCH * DCH;                            // 512
    float* SUMSQ = KSSUM + HD;                                         // 2

    const int nX8 = (NN * INCH) / 8;
    const int nW8 = (HD * INCH) / 8;
    cvt_kernel<<<2048, 256, 0, stream>>>(q_in, XQ, nX8);
    cvt_kernel<<<2048, 256, 0, stream>>>(s_in, XS, nX8);
    cvt_kernel<<<128, 256, 0, stream>>>(Wq_w, WQB, nW8);
    cvt_kernel<<<128, 256, 0, stream>>>(Wk_w, WKB, nW8);
    cvt_kernel<<<128, 256, 0, stream>>>(Wv_w, WVB, nW8);
    zero_kernel<<<131, 256, 0, stream>>>(KVS, HEADS * DCH * DCH + HD + 2);

    proj_kernel<<<NN / 16, 256, 0, stream>>>(XQ, WQB, Wq_b, QS, &SUMSQ[0], nullptr);
    proj_kernel<<<NN / 16, 256, 0, stream>>>(XS, WKB, Wk_b, KS, &SUMSQ[1], KSSUM);
    proj_kernel<<<NN / 16, 256, 0, stream>>>(XS, WVB, Wv_b, VS, nullptr, nullptr);

    kvs_kernel<<<dim3(64, 8), 256, 0, stream>>>(KS, VS, KVS);
    attn_kernel<<<512, 256, 0, stream>>>(QS, VS, KVS, KSSUM, SUMSQ, (float*)d_out);
}
